// MultiHeadAttention_4483945857431
// MI455X (gfx1250) — compile-verified, matrix operand order checked by reading, not run on hardware
//
#include <hip/hip_runtime.h>
#include <hip/hip_bf16.h>

// ---------------------------------------------------------------------------
// MultiHeadAttention for MI455X (gfx1250, wave32, WMMA).
// All GEMM work (QKV proj, Q.K^T, P.V, out proj) runs on
// v_wmma_f32_16x16x32_bf16 with f32 accumulation.
// Round 4: K-only cross-iteration double buffering (V loads are covered by
// the softmax inside the iteration) to get back under 256 VGPRs and kill the
// s_set_vgpr_msb churn; 1/sqrt(dk) folded into the Q projection epilogue.
// ---------------------------------------------------------------------------

typedef __bf16 bf16_t;
typedef __attribute__((ext_vector_type(8)))  __bf16 v8bf;
typedef __attribute__((ext_vector_type(16))) __bf16 v16bf;
typedef __attribute__((ext_vector_type(8)))  float  v8f;

#define B_DIM  2
#define HEADS  16
#define SEQ    2048
#define DMODEL 1024
#define DK     64

// A/B fragment loader.  Per ISA 16-bit A layout (wave32):
//   lane l holds row (l&15); element e holds K = kbase + (e>>3)*16 + (l>>4)*8 + (e&7)
// -> two contiguous 16-byte chunks per lane.  B mirrors A with lane<->column.
static __device__ __forceinline__ v16bf load_frag(const bf16_t* __restrict__ base,
                                                  int row, int stride, int kbase, int hi) {
  const bf16_t* p = base + (size_t)row * (size_t)stride + kbase + hi * 8;
  v8bf lo  = *(const v8bf*)(p);
  v8bf hi8 = *(const v8bf*)(p + 16);
  return __builtin_shufflevector(lo, hi8, 0,1,2,3,4,5,6,7,8,9,10,11,12,13,14,15);
}

static __device__ __forceinline__ v8f wmma32(v16bf a, v16bf b, v8f c) {
  // (neg_a, A, neg_b, B, c_mod, C, reuse_a, reuse_b)
  return __builtin_amdgcn_wmma_f32_16x16x32_bf16(false, a, false, b, (short)0, c,
                                                 false, false);
}

// ---------------------------------------------------------------------------
// Stage 1: fp32 -> bf16 conversion / weight transpose
// ---------------------------------------------------------------------------
__global__ void mha_cvt_bf16(const float* __restrict__ src, bf16_t* __restrict__ dst, int n) {
  int i = blockIdx.x * blockDim.x + threadIdx.x;
  if (i < n) dst[i] = (bf16_t)src[i];
}

// Wt[n][k] = W[k][n]  (1024x1024), writes are contiguous
__global__ void mha_transpose_bf16(const float* __restrict__ W, bf16_t* __restrict__ Wt) {
  int i = blockIdx.x * blockDim.x + threadIdx.x;   // i = n*1024 + k
  int n = i >> 10, k = i & 1023;
  Wt[i] = (bf16_t)W[k * DMODEL + n];
}

// ---------------------------------------------------------------------------
// Stage 2/5: bf16 GEMM  C[M,N] = (A[M,K] * Bt[N,K]^T + bias) * scale, K=1024.
// Each wave computes a 32x64 output block: 2 A fragments x 4 B fragments,
// 8 accumulators; every B fragment is consumed by two WMMAs.  k-loop is
// unrolled 2x so next-step loads issue under current WMMAs.
// MODE 0: f32 row-major (final output)
// MODE 1: bf16 head-split  [b*H+h][s][dk]        (Q with scale=1/8, K)
// MODE 2: bf16 head-split transposed [b*H+h][dk][s]  (V)
// ---------------------------------------------------------------------------
template<int MODE>
__global__ void mha_gemm_bf16(const bf16_t* __restrict__ A, const bf16_t* __restrict__ Bt,
                              const float* __restrict__ bias, void* __restrict__ outp,
                              int M, int N, float scale) {
  const int lane = threadIdx.x & 31;
  const int hi   = lane >> 4;
  const int ml   = lane & 15;
  const int wid  = (int)((blockIdx.x * blockDim.x + threadIdx.x) >> 5);
  const int n64  = N >> 6;
  const int mt   = wid / n64;          // 32-row tile index
  const int nt   = wid % n64;          // 64-col tile index

  v8f acc[2][4] = {};
  const int arow0 = mt * 32 + ml;
  const int arow1 = arow0 + 16;

#pragma unroll 2
  for (int k = 0; k < DMODEL; k += 32) {
    v16bf a0 = load_frag(A, arow0, DMODEL, k, hi);
    v16bf a1 = load_frag(A, arow1, DMODEL, k, hi);
#pragma unroll
    for (int j = 0; j < 4; ++j) {
      v16bf b = load_frag(Bt, nt * 64 + j * 16 + ml, DMODEL, k, hi);
      acc[0][j] = wmma32(a0, b, acc[0][j]);
      acc[1][j] = wmma32(a1, b, acc[1][j]);
    }
  }

#pragma unroll
  for (int half = 0; half < 2; ++half) {
#pragma unroll
    for (int j = 0; j < 4; ++j) {
#pragma unroll
      for (int r = 0; r < 8; ++r) {
        int m = mt * 32 + half * 16 + r + 8 * hi;  // C layout: VGPR r -> row r+8*(l>>4)
        int n = nt * 64 + j * 16 + ml;             //           column = lane&15
        float v = (acc[half][j][r] + bias[n]) * scale;
        if (MODE == 0) {
          ((float*)outp)[(size_t)m * N + n] = v;
        } else {
          int b = m >> 11, s = m & (SEQ - 1);
          int h = n >> 6,  d = n & (DK - 1);
          if (MODE == 1)
            ((bf16_t*)outp)[((size_t)(b * HEADS + h) * SEQ + s) * DK + d] = (bf16_t)v;
          else
            ((bf16_t*)outp)[((size_t)(b * HEADS + h) * DK + d) * SEQ + s] = (bf16_t)v;
        }
      }
    }
  }
}

// ---------------------------------------------------------------------------
// Stage 3: flash attention.  One wave per (b,h, 32-query-row block).
// Online softmax; per-row reductions are half-wave shfl_xor (wave32).
// Probability tiles are re-laid out C-fragment -> A-fragment through 2KB LDS.
// K fragments are double-buffered across iterations (consumed at loop head);
// V fragments are loaded inside the iteration before the softmax, whose
// ~150 VALU/DS instructions cover their latency.  Q is pre-scaled by 1/8.
// ---------------------------------------------------------------------------
__global__ void __launch_bounds__(32)
mha_attn(const bf16_t* __restrict__ Qh, const bf16_t* __restrict__ Kh,
         const bf16_t* __restrict__ Vt, bf16_t* __restrict__ ctx) {
  __shared__ bf16_t pls[32 * 32];

  const int lane = threadIdx.x & 31;
  const int hi = lane >> 4, ml = lane & 15;
  const int qblocks = SEQ / 32;
  const int bh = blockIdx.x / qblocks;
  const int qb = blockIdx.x % qblocks;

  const bf16_t* Qp = Qh + (size_t)bh * SEQ * DK;
  const bf16_t* Kp = Kh + (size_t)bh * SEQ * DK;
  const bf16_t* Vp = Vt + (size_t)bh * DK * SEQ;

  // Q fragments: 2 query halves x 2 d-chunks
  v16bf a[2][2];
#pragma unroll
  for (int qh = 0; qh < 2; ++qh) {
    a[qh][0] = load_frag(Qp, qb * 32 + qh * 16 + ml, DK, 0,  hi);
    a[qh][1] = load_frag(Qp, qb * 32 + qh * 16 + ml, DK, 32, hi);
  }

  float mstate[2][8], lstate[2][8];
#pragma unroll
  for (int qh = 0; qh < 2; ++qh)
#pragma unroll
    for (int r = 0; r < 8; ++r) { mstate[qh][r] = -1e30f; lstate[qh][r] = 0.f; }
  v8f acc[2][4] = {};   // ctx 32x64 = 2 q-halves x four 16x16 f32 tiles

  // prologue: K fragments for tile 0
  v16bf kb0l = load_frag(Kp, ml,      DK, 0,  hi);
  v16bf kb0h = load_frag(Kp, ml,      DK, 32, hi);
  v16bf kb1l = load_frag(Kp, 16 + ml, DK, 0,  hi);
  v16bf kb1h = load_frag(Kp, 16 + ml, DK, 32, hi);

#pragma unroll 2
  for (int k0 = 0; k0 < SEQ; k0 += 32) {
    v8f zero = {};
    // scores for current tile (consumes kb*; Q pre-scaled by 1/sqrt(dk))
    v8f s[2][2];
#pragma unroll
    for (int qh = 0; qh < 2; ++qh) {
      s[qh][0] = wmma32(a[qh][1], kb0h, zero);
      s[qh][0] = wmma32(a[qh][0], kb0l, s[qh][0]);
      s[qh][1] = wmma32(a[qh][1], kb1h, zero);
      s[qh][1] = wmma32(a[qh][0], kb1l, s[qh][1]);
    }

    // issue next tile's K loads (row clamped -> branch-free) and this tile's
    // V loads; both complete under the softmax below.
    const int kn = (k0 + 32 < SEQ) ? (k0 + 32) : 0;
    v16bf nk0l = load_frag(Kp, kn + ml,      DK, 0,  hi);
    v16bf nk0h = load_frag(Kp, kn + ml,      DK, 32, hi);
    v16bf nk1l = load_frag(Kp, kn + 16 + ml, DK, 0,  hi);
    v16bf nk1h = load_frag(Kp, kn + 16 + ml, DK, 32, hi);
    v16bf vb[4];
#pragma unroll
    for (int j = 0; j < 4; ++j)
      vb[j] = load_frag(Vp, j * 16 + ml, SEQ, k0, hi);

    __syncthreads();   // protect pls against previous iteration's readers
#pragma unroll
    for (int qh = 0; qh < 2; ++qh) {
      float p0[8], p1[8], corr[8];
#pragma unroll
      for (int r = 0; r < 8; ++r) {
        float x0 = s[qh][0][r];
        float x1 = s[qh][1][r];
        // row max over 32 key cols: pair max + 16-lane xor reduction
        float t = fmaxf(x0, x1);
        t = fmaxf(t, __shfl_xor(t, 1, 32));
        t = fmaxf(t, __shfl_xor(t, 2, 32));
        t = fmaxf(t, __shfl_xor(t, 4, 32));
        t = fmaxf(t, __shfl_xor(t, 8, 32));
        float nm = fmaxf(mstate[qh][r], t);
        corr[r] = __expf(mstate[qh][r] - nm);
        mstate[qh][r] = nm;
        p0[r] = __expf(x0 - nm);
        p1[r] = __expf(x1 - nm);
        float rs = p0[r] + p1[r];
        rs += __shfl_xor(rs, 1, 32);
        rs += __shfl_xor(rs, 2, 32);
        rs += __shfl_xor(rs, 4, 32);
        rs += __shfl_xor(rs, 8, 32);
        lstate[qh][r] = lstate[qh][r] * corr[r] + rs;
      }
#pragma unroll
      for (int j = 0; j < 4; ++j)
#pragma unroll
        for (int r = 0; r < 8; ++r)
          acc[qh][j][r] *= corr[r];

      // C-fragment -> A-fragment relayout of probabilities via LDS
#pragma unroll
      for (int r = 0; r < 8; ++r) {
        int mrow = qh * 16 + r + 8 * hi;
        pls[mrow * 32 + ml]      = (bf16_t)p0[r];
        pls[mrow * 32 + 16 + ml] = (bf16_t)p1[r];
      }
    }
    __syncthreads();
    v16bf pa0 = load_frag(pls,           ml, 32, 0, hi);  // P rows 0..15  as A frag
    v16bf pa1 = load_frag(pls + 16 * 32, ml, 32, 0, hi);  // P rows 16..31 as A frag

    // ctx += P(32x32) @ V(32x64): Vt rows are features -> B-fragment rows
#pragma unroll
    for (int j = 0; j < 4; ++j) {
      acc[0][j] = wmma32(pa0, vb[j], acc[0][j]);
      acc[1][j] = wmma32(pa1, vb[j], acc[1][j]);
    }

    // rotate K double buffer (unroll-2 turns these into pure renaming)
    kb0l = nk0l; kb0h = nk0h; kb1l = nk1l; kb1h = nk1h;
  }

  // normalize and store concatenated heads: ctx[b][s][h*64+d] (bf16)
  const int b = bh / HEADS, h = bh % HEADS;
#pragma unroll
  for (int qh = 0; qh < 2; ++qh) {
#pragma unroll
    for (int r = 0; r < 8; ++r) {
      float inv = 1.f / lstate[qh][r];
      int s = qb * 32 + qh * 16 + r + 8 * hi;
#pragma unroll
      for (int j = 0; j < 4; ++j) {
        int d = j * 16 + ml;
        ctx[(size_t)(b * SEQ + s) * DMODEL + h * DK + d] = (bf16_t)(acc[qh][j][r] * inv);
      }
    }
  }
}

// ---------------------------------------------------------------------------
extern "C" void kernel_launch(void* const* d_in, const int* in_sizes, int n_in,
                              void* d_out, int out_size, void* d_ws, size_t ws_size,
                              hipStream_t stream) {
  (void)in_sizes; (void)n_in; (void)out_size; (void)ws_size;

  const float* q  = (const float*)d_in[0];
  const float* k  = (const float*)d_in[1];
  const float* v  = (const float*)d_in[2];
  const float* Wq = (const float*)d_in[3];
  const float* bq = (const float*)d_in[4];
  const float* Wk = (const float*)d_in[5];
  const float* bk = (const float*)d_in[6];
  const float* Wv = (const float*)d_in[7];
  const float* bv = (const float*)d_in[8];
  const float* Wo = (const float*)d_in[9];
  const float* bo = (const float*)d_in[10];

  char* ws = (char*)d_ws;
  const size_t SZ_X = (size_t)B_DIM * SEQ * DMODEL * sizeof(bf16_t); // 8 MB
  const size_t SZ_W = (size_t)DMODEL * DMODEL * sizeof(bf16_t);      // 2 MB
  bf16_t* xq  = (bf16_t*)(ws);
  bf16_t* xk  = (bf16_t*)(ws + SZ_X);
  bf16_t* xv  = (bf16_t*)(ws + 2 * SZ_X);
  bf16_t* WqT = (bf16_t*)(ws + 3 * SZ_X);
  bf16_t* WkT = (bf16_t*)(ws + 3 * SZ_X + SZ_W);
  bf16_t* WvT = (bf16_t*)(ws + 3 * SZ_X + 2 * SZ_W);
  bf16_t* WoT = (bf16_t*)(ws + 3 * SZ_X + 3 * SZ_W);
  bf16_t* Qh  = (bf16_t*)(ws + 3 * SZ_X + 4 * SZ_W);
  bf16_t* Kh  = (bf16_t*)(ws + 4 * SZ_X + 4 * SZ_W);
  bf16_t* Vt  = (bf16_t*)(ws + 5 * SZ_X + 4 * SZ_W);
  bf16_t* ctx = (bf16_t*)(ws + 6 * SZ_X + 4 * SZ_W);   // total 64 MB

  const int NX = B_DIM * SEQ * DMODEL;   // 4,194,304
  mha_cvt_bf16<<<NX / 256, 256, 0, stream>>>(q, xq, NX);
  mha_cvt_bf16<<<NX / 256, 256, 0, stream>>>(k, xk, NX);
  mha_cvt_bf16<<<NX / 256, 256, 0, stream>>>(v, xv, NX);

  const int NW = DMODEL * DMODEL;
  mha_transpose_bf16<<<NW / 256, 256, 0, stream>>>(Wq, WqT);
  mha_transpose_bf16<<<NW / 256, 256, 0, stream>>>(Wk, WkT);
  mha_transpose_bf16<<<NW / 256, 256, 0, stream>>>(Wv, WvT);
  mha_transpose_bf16<<<NW / 256, 256, 0, stream>>>(Wo, WoT);

  const int M = B_DIM * SEQ, N = DMODEL;
  const int gemm_blocks = (M / 32) * (N / 64) / 4;   // 4 waves (128 thr) per block
  // Q projection carries the 1/sqrt(dk) attention scale
  mha_gemm_bf16<1><<<gemm_blocks, 128, 0, stream>>>(xq, WqT, bq, Qh, M, N, 0.125f);
  mha_gemm_bf16<1><<<gemm_blocks, 128, 0, stream>>>(xk, WkT, bk, Kh, M, N, 1.0f);
  mha_gemm_bf16<2><<<gemm_blocks, 128, 0, stream>>>(xv, WvT, bv, Vt, M, N, 1.0f);

  mha_attn<<<B_DIM * HEADS * (SEQ / 32), 32, 0, stream>>>(Qh, Kh, Vt, ctx);

  mha_gemm_bf16<0><<<gemm_blocks, 128, 0, stream>>>(ctx, WoT, bo, d_out, M, N, 1.0f);
}
